// GrapherModule_15942918603267
// MI455X (gfx1250) — matrix operand order repaired and checked by reading, not compiled, not run on hardware
//
#include <hip/hip_runtime.h>
#include <hip/hip_bf16.h>
#include <math.h>
#include <stdint.h>

// ---------------------------------------------------------------------------
// GrapherModule (ViG block) for MI455X / gfx1250, wave32, WMMA bf16->f32.
//   B=32, C=192, HC=384, N=H*W=1024, K=9, R=B*N=32768
// Round 2 changes:
//  * Weights repacked once into WMMA B-fragment lane layout -> inner loop is
//    pure b128 loads + v_wmma (no scalar u16 gathers).
//  * Packed B panel staged to LDS via global_load_async_to_lds_b128 (+
//    s_wait_asynccnt), read back as ds_load_b128.
//  * 2 column tiles per wave (one A fragment feeds two WMMAs).
//  * Coalesced two-stage BN statistics.
// ---------------------------------------------------------------------------

#define B_   32
#define C_   192
#define HC_  384
#define N_   1024
#define K_   9
#define R_   (B_ * N_)
#define EPS_ 1e-5f
#define MAXKT32 12        // max Kt/32 among the GEMMs (Kt=384)
#define STAT_CHUNKS 32

typedef __attribute__((ext_vector_type(16))) __bf16 v16bf;
typedef __attribute__((ext_vector_type(8)))  float  v8f;

// ---------------------------------------------------------------------------
// WMMA fragment loaders (wave32); A and B share this code's contiguous k order.
// A 16x32 bf16: lane<16 -> M=lane, K=0..15; lane>=16 -> M=lane-16,
//   K=16..31  => per lane one contiguous 32B run (vectorizes).
// B 32x16 bf16: lane<16 -> N=lane, K=0..15; lane>=16 -> N=lane-16, K=16..31.
// C/D 16x16 f32: VGPR r -> (M=r,N=lane) lo / (M=r+8,N=lane-16) hi.
// ---------------------------------------------------------------------------
__device__ inline v16bf load_a_frag(const __bf16* A, int lda, int row0, int k0,
                                    int lane) {
  const int half = lane >> 4, m = lane & 15;
  const __bf16* base = A + (size_t)(row0 + m) * lda + k0;
  v16bf a;
#pragma unroll
  for (int e = 0; e < 16; ++e) {
    int v = e >> 1, r = e & 1;
    int K = (half ? 16 : 0) + 2 * v + r;
    a[e] = base[K];
  }
  return a;
}

// B = Y^T tiles for the distance GEMM: B[k,n] = Y[col0+n, k]; per lane the 16
// bf16 are contiguous in memory (vectorizes to 2x b128).
__device__ inline v16bf load_bT_frag(const __bf16* Y, int ldy, int k0, int col0,
                                     int lane) {
  const int half = lane >> 4, n = lane & 15;
  const __bf16* base = Y + (size_t)(col0 + n) * ldy + k0 + (half ? 16 : 0);
  v16bf b;
#pragma unroll
  for (int e = 0; e < 16; ++e) b[e] = base[e];
  return b;
}

// ---------------------------------------------------------------------------
// Repack weight W [Kt x Nc] (f32 row-major) into B-fragment order:
//   Bpk[((nt*(Kt/32)+kt)*32 + lane)*16 + e] = bf16( W[(kt*32+K) * Nc + nt*16+n] )
// with half=lane>>4, n=lane&15, K=(half?16:0)+e.  One thread per element.
// ---------------------------------------------------------------------------
__global__ void pack_b_frag(const float* __restrict__ W,
                            __bf16* __restrict__ Bpk, int Kt, int Nc) {
  int t = blockIdx.x * blockDim.x + threadIdx.x;
  if (t >= Kt * Nc) return;
  const int KT32 = Kt >> 5;
  int e    = t & 15;
  int lane = (t >> 4) & 31;
  int tile = t >> 9;            // nt*KT32 + kt
  int kt   = tile % KT32;
  int nt   = tile / KT32;
  int half = lane >> 4, n = lane & 15;
  int K = (half ? 16 : 0) + e;
  Bpk[t] = (__bf16)W[(size_t)(kt * 32 + K) * Nc + (nt * 16 + n)];
}

// ---------------------------------------------------------------------------
// GEMM: C[M x Nc] = A[M x Kt] @ W[Kt x Nc] + bias, W pre-packed (Bpk).
// blockDim (32,4): 4 waves, each computes a 16-row x 32-col slab (2 tiles).
// Packed-B panel for this block's 2 column tiles is staged into LDS with
// CDNA5 async copies, then read as b128 fragments.
// ---------------------------------------------------------------------------
__global__ void wmma_gemm_bf16(const __bf16* __restrict__ A,
                               const __bf16* __restrict__ Bpk,
                               const float* __restrict__ bias,
                               float* __restrict__ C, int M, int Nc, int Kt) {
  __shared__ alignas(32) __bf16 sB[2 * MAXKT32 * 512];
  const int KT32 = Kt >> 5;
  const int tid  = threadIdx.y * 32 + threadIdx.x;
  const int nt0  = blockIdx.x * 2;

  // ---- async stage: 2 * KT32 * 1KB panel, 16B per transfer per lane ----
  const __bf16* panel = Bpk + (size_t)nt0 * KT32 * 512;
  const int chunks = KT32 * 128;                 // (2*KT32*1024)/16
  const uint32_t ldsBase = (uint32_t)(uintptr_t)(&sB[0]);
  for (int tc = tid; tc < chunks; tc += 128) {   // uniform trip count
    uint64_t gaddr = (uint64_t)(uintptr_t)panel + (uint64_t)tc * 16;
    uint32_t laddr = ldsBase + (uint32_t)tc * 16;
    asm volatile("global_load_async_to_lds_b128 %0, %1, off"
                 :: "v"(laddr), "v"(gaddr) : "memory");
  }
  asm volatile("s_wait_asynccnt 0x0" ::: "memory");
  __syncthreads();

  const int lane = threadIdx.x;
  const int row0 = (blockIdx.y * 4 + threadIdx.y) * 16;

  v8f acc0 = {}, acc1 = {};
  for (int kt = 0; kt < KT32; ++kt) {
    const int k0 = kt * 32;
    if (kt + 2 <= KT32) {  // uniform; prefetch next A slab into GL2
      __builtin_prefetch(A + (size_t)(row0 + (lane & 15)) * Kt + k0 + 32, 0, 1);
    }
    v16bf a  = load_a_frag(A, Kt, row0, k0, lane);
    v16bf b0 = *(const v16bf*)&sB[((size_t)kt * 512) + (size_t)lane * 16];
    v16bf b1 = *(const v16bf*)&sB[((size_t)(KT32 + kt) * 512) + (size_t)lane * 16];
    acc0 = __builtin_amdgcn_wmma_f32_16x16x32_bf16(false, a, false, b0,
                                                   (short)0, acc0, false, false);
    acc1 = __builtin_amdgcn_wmma_f32_16x16x32_bf16(false, a, false, b1,
                                                   (short)0, acc1, false, false);
  }

  const int half = lane >> 4, n = lane & 15;
  const int colA = nt0 * 16 + n;
  const int colB = colA + 16;
  const float bvA = bias[colA], bvB = bias[colB];
#pragma unroll
  for (int r = 0; r < 8; ++r) {
    int crow = row0 + r + (half ? 8 : 0);
    C[(size_t)crow * Nc + colA] = acc0[r] + bvA;
    C[(size_t)crow * Nc + colB] = acc1[r] + bvB;
  }
}

// dist[n,m] = x2[n] + x2[m] - 2 * (Yb @ Yb^T)[n,m]  for one batch.
__global__ void wmma_dist_bf16(const __bf16* __restrict__ Yb,
                               const float* __restrict__ x2b,
                               float* __restrict__ dist) {
  const int lane = threadIdx.x;
  const int col0 = blockIdx.x * 16;
  const int row0 = (blockIdx.y * 4 + threadIdx.y) * 16;

  v8f acc = {};
#pragma unroll
  for (int k0 = 0; k0 < C_; k0 += 32) {
    v16bf a = load_a_frag(Yb, C_, row0, k0, lane);
    v16bf b = load_bT_frag(Yb, C_, k0, col0, lane);
    acc = __builtin_amdgcn_wmma_f32_16x16x32_bf16(false, a, false, b, (short)0,
                                                  acc, false, false);
  }
  const int half = lane >> 4, n = lane & 15;
  const int ccol = col0 + n;
  const float xm = x2b[ccol];
#pragma unroll
  for (int r = 0; r < 8; ++r) {
    int crow = row0 + r + (half ? 8 : 0);
    dist[(size_t)crow * N_ + ccol] = x2b[crow] + xm - 2.0f * acc[r];
  }
}

// ---------------------------------------------------------------------------
// Elementwise / reduction kernels
// ---------------------------------------------------------------------------
// x [B,C,N] -> Xbf [(B*N) x C] bf16 (transpose to node-major).
__global__ void transpose_x_bf16(const float* __restrict__ x,
                                 __bf16* __restrict__ xbf) {
  int t = blockIdx.x * blockDim.x + threadIdx.x;
  if (t >= R_ * C_) return;
  int r = t / C_, c = t - r * C_;
  int b = r / N_, n = r - b * N_;
  xbf[t] = (__bf16)x[(size_t)b * C_ * N_ + (size_t)c * N_ + n];
}

// Stage 1: coalesced partial sums. blockDim 256 = 32 channels x 8 rows.
// grid = (Ch/32, STAT_CHUNKS).
__global__ void bn_stats_partial(const float* __restrict__ X,
                                 float* __restrict__ psum,
                                 float* __restrict__ psmq, int R, int Ch) {
  __shared__ float s1[256];
  __shared__ float s2[256];
  const int ch = blockIdx.x * 32 + (threadIdx.x & 31);
  const int rl = threadIdx.x >> 5;                 // 0..7
  const int rows = R / STAT_CHUNKS;
  const int r0 = blockIdx.y * rows;
  float s = 0.f, ss = 0.f;
  for (int r = r0 + rl; r < r0 + rows; r += 8) {
    float v = X[(size_t)r * Ch + ch];
    s += v;
    ss += v * v;
  }
  s1[threadIdx.x] = s;
  s2[threadIdx.x] = ss;
  __syncthreads();
  for (int off = 128; off >= 32; off >>= 1) {
    if (threadIdx.x < off) {
      s1[threadIdx.x] += s1[threadIdx.x + off];
      s2[threadIdx.x] += s2[threadIdx.x + off];
    }
    __syncthreads();
  }
  if (threadIdx.x < 32) {
    psum[(size_t)blockIdx.y * Ch + ch] = s1[threadIdx.x];
    psmq[(size_t)blockIdx.y * Ch + ch] = s2[threadIdx.x];
  }
}

// Stage 2: fold STAT_CHUNKS partials into mean/var.
__global__ void bn_stats_final(const float* __restrict__ psum,
                               const float* __restrict__ psmq,
                               float* __restrict__ mean,
                               float* __restrict__ var, int Ch, int R) {
  int ch = blockIdx.x * blockDim.x + threadIdx.x;
  if (ch >= Ch) return;
  float s = 0.f, ss = 0.f;
  for (int i = 0; i < STAT_CHUNKS; ++i) {
    s += psum[(size_t)i * Ch + ch];
    ss += psmq[(size_t)i * Ch + ch];
  }
  float m = s / (float)R;
  mean[ch] = m;
  var[ch] = ss / (float)R - m * m;
}

// y = BN(pre) -> f32 + bf16 copies.
__global__ void bn_apply_y(const float* __restrict__ pre,
                           const float* __restrict__ mean,
                           const float* __restrict__ var,
                           const float* __restrict__ g,
                           const float* __restrict__ be,
                           float* __restrict__ y, __bf16* __restrict__ ybf) {
  int t = blockIdx.x * blockDim.x + threadIdx.x;
  if (t >= R_ * C_) return;
  int c = t % C_;
  float val = (pre[t] - mean[c]) * rsqrtf(var[c] + EPS_) * g[c] + be[c];
  y[t] = val;
  ybf[t] = (__bf16)val;
}

// hbf = bf16( gelu_exact( BN(pre) ) ), pre [R x HC].
__global__ void bn_gelu_h(const float* __restrict__ pre,
                          const float* __restrict__ mean,
                          const float* __restrict__ var,
                          const float* __restrict__ g,
                          const float* __restrict__ be,
                          __bf16* __restrict__ hbf) {
  int t = blockIdx.x * blockDim.x + threadIdx.x;
  if (t >= R_ * HC_) return;
  int c = t % HC_;
  float val = (pre[t] - mean[c]) * rsqrtf(var[c] + EPS_) * g[c] + be[c];
  val = 0.5f * val * (1.0f + erff(val * 0.70710678118654752f));
  hbf[t] = (__bf16)val;
}

// out[b,c,n] = BN(pre)[r,c] + x[b,c,n]   (residual, back to [B,C,H,W]).
__global__ void bn_final(const float* __restrict__ pre,
                         const float* __restrict__ mean,
                         const float* __restrict__ var,
                         const float* __restrict__ g,
                         const float* __restrict__ be,
                         const float* __restrict__ xin,
                         float* __restrict__ out) {
  int t = blockIdx.x * blockDim.x + threadIdx.x;
  if (t >= R_ * C_) return;
  int r = t / C_, c = t - r * C_;
  int b = r / N_, n = r - b * N_;
  float val = (pre[t] - mean[c]) * rsqrtf(var[c] + EPS_) * g[c] + be[c];
  size_t o = (size_t)b * C_ * N_ + (size_t)c * N_ + n;
  out[o] = val + xin[o];
}

// x2[r] = sum_c y[r,c]^2
__global__ void row_norms(const float* __restrict__ y, float* __restrict__ x2) {
  int r = blockIdx.x * blockDim.x + threadIdx.x;
  if (r >= R_) return;
  float s = 0.f;
  const float* row = y + (size_t)r * C_;
  for (int c = 0; c < C_; ++c) s += row[c] * row[c];
  x2[r] = s;
}

// top-9 smallest per row; dist is symmetric so thread n walks column n
// (dist[m*N+n]) for coalesced access.
__global__ void topk9(const float* __restrict__ dist, int* __restrict__ idxb) {
  int n = blockIdx.x * blockDim.x + threadIdx.x;
  if (n >= N_) return;
  float bd[K_];
  int bi[K_];
#pragma unroll
  for (int k = 0; k < K_; ++k) {
    bd[k] = 3.0e38f;
    bi[k] = 0;
  }
  for (int m = 0; m < N_; ++m) {
    float d = dist[(size_t)m * N_ + n];
    if (d < bd[K_ - 1]) {
      int p = K_ - 1;
      while (p > 0 && bd[p - 1] > d) {
        bd[p] = bd[p - 1];
        bi[p] = bi[p - 1];
        --p;
      }
      bd[p] = d;
      bi[p] = m;
    }
  }
#pragma unroll
  for (int k = 0; k < K_; ++k) idxb[(size_t)n * K_ + k] = bi[k];
}

// agg = max_k (y[idx_k] - y_i); hcat = [ y | agg ] as bf16 [R x 2C].
__global__ void mr_aggregate(const float* __restrict__ y,
                             const int* __restrict__ idx,
                             __bf16* __restrict__ hcat) {
  int t = blockIdx.x * blockDim.x + threadIdx.x;
  if (t >= R_ * C_) return;
  int r = t / C_, c = t - r * C_;
  int b = r / N_;
  float yi = y[(size_t)r * C_ + c];
  float mx = -3.0e38f;
  const int base = b * N_;
#pragma unroll
  for (int k = 0; k < K_; ++k) {
    int j = idx[(size_t)r * K_ + k];
    float v = y[((size_t)(base + j)) * C_ + c] - yi;
    mx = fmaxf(mx, v);
  }
  hcat[(size_t)r * (2 * C_) + c] = (__bf16)yi;
  hcat[(size_t)r * (2 * C_) + C_ + c] = (__bf16)mx;
}

// ---------------------------------------------------------------------------
// Host-side orchestration
// ---------------------------------------------------------------------------
extern "C" void kernel_launch(void* const* d_in, const int* in_sizes, int n_in,
                              void* d_out, int out_size, void* d_ws,
                              size_t ws_size, hipStream_t stream) {
  (void)in_sizes; (void)n_in; (void)out_size; (void)ws_size;
  const float* x   = (const float*)d_in[0];
  const float* W1  = (const float*)d_in[1];
  const float* b1  = (const float*)d_in[2];
  const float* g1  = (const float*)d_in[3];
  const float* be1 = (const float*)d_in[4];
  const float* Wg  = (const float*)d_in[5];
  const float* bg  = (const float*)d_in[6];
  const float* gg  = (const float*)d_in[7];
  const float* beg = (const float*)d_in[8];
  const float* W2  = (const float*)d_in[9];
  const float* b2  = (const float*)d_in[10];
  const float* g2  = (const float*)d_in[11];
  const float* be2 = (const float*)d_in[12];
  float* out = (float*)d_out;

  // Workspace carve-up (256B aligned).
  char* ws = (char*)d_ws;
  size_t off = 0;
  auto carve = [&](size_t bytes) {
    char* p = ws + off;
    off = (off + bytes + 255) & ~(size_t)255;
    return p;
  };
  __bf16* w1pk  = (__bf16*)carve((size_t)C_ * C_ * 2);
  __bf16* wgpk  = (__bf16*)carve((size_t)2 * C_ * HC_ * 2);
  __bf16* w2pk  = (__bf16*)carve((size_t)HC_ * C_ * 2);
  __bf16* xbf   = (__bf16*)carve((size_t)R_ * C_ * 2);
  float*  pre   = (float*) carve((size_t)R_ * HC_ * 4);  // reused 3x
  float*  y     = (float*) carve((size_t)R_ * C_ * 4);
  __bf16* ybf   = (__bf16*)carve((size_t)R_ * C_ * 2);
  float*  x2    = (float*) carve((size_t)R_ * 4);
  int*    idx   = (int*)   carve((size_t)R_ * K_ * 4);
  float*  dist  = (float*) carve((size_t)N_ * N_ * 4);   // per-batch, reused
  __bf16* hcat  = (__bf16*)carve((size_t)R_ * 2 * C_ * 2);
  __bf16* hbf   = (__bf16*)carve((size_t)R_ * HC_ * 2);
  float*  mean  = (float*) carve((size_t)HC_ * 4);
  float*  var   = (float*) carve((size_t)HC_ * 4);
  float*  psum  = (float*) carve((size_t)STAT_CHUNKS * HC_ * 4);
  float*  psmq  = (float*) carve((size_t)STAT_CHUNKS * HC_ * 4);

  const dim3 wblk(32, 4);  // 4 wave32 waves per workgroup
  const int TPB = 256;
  auto ceil_div = [](int a, int b) { return (a + b - 1) / b; };

  auto run_stats = [&](const float* X, int Ch) {
    bn_stats_partial<<<dim3(Ch / 32, STAT_CHUNKS), 256, 0, stream>>>(X, psum, psmq, R_, Ch);
    bn_stats_final<<<ceil_div(Ch, 128), 128, 0, stream>>>(psum, psmq, mean, var, Ch, R_);
  };

  // 1) weight repack (fragment layout) + x transpose/convert
  pack_b_frag<<<ceil_div(C_ * C_, TPB), TPB, 0, stream>>>(W1, w1pk, C_, C_);
  pack_b_frag<<<ceil_div(2 * C_ * HC_, TPB), TPB, 0, stream>>>(Wg, wgpk, 2 * C_, HC_);
  pack_b_frag<<<ceil_div(HC_ * C_, TPB), TPB, 0, stream>>>(W2, w2pk, HC_, C_);
  transpose_x_bf16<<<ceil_div(R_ * C_, TPB), TPB, 0, stream>>>(x, xbf);

  // 2) fc1 GEMM: pre = Xbf @ W1 + b1   [R x C]
  wmma_gemm_bf16<<<dim3(C_ / 32, R_ / 64), wblk, 0, stream>>>(xbf, w1pk, b1, pre, R_, C_, C_);

  // 3) BN -> y / ybf
  run_stats(pre, C_);
  bn_apply_y<<<ceil_div(R_ * C_, TPB), TPB, 0, stream>>>(pre, mean, var, g1, be1, y, ybf);

  // 4) row norms
  row_norms<<<ceil_div(R_, TPB), TPB, 0, stream>>>(y, x2);

  // 5) per-batch kNN (stream-serialized reuse of `dist`)
  for (int b = 0; b < B_; ++b) {
    const __bf16* Yb = ybf + (size_t)b * N_ * C_;
    const float* x2b = x2 + (size_t)b * N_;
    wmma_dist_bf16<<<dim3(N_ / 16, N_ / 64), wblk, 0, stream>>>(Yb, x2b, dist);
    topk9<<<ceil_div(N_, TPB), TPB, 0, stream>>>(dist, idx + (size_t)b * N_ * K_);
  }

  // 6) MRConv aggregate + concat
  mr_aggregate<<<ceil_div(R_ * C_, TPB), TPB, 0, stream>>>(y, idx, hcat);

  // 7) graph conv GEMM: pre = hcat @ Wg + bg   [R x HC]
  wmma_gemm_bf16<<<dim3(HC_ / 32, R_ / 64), wblk, 0, stream>>>(hcat, wgpk, bg, pre, R_, HC_, 2 * C_);

  // 8) BN + GELU -> hbf
  run_stats(pre, HC_);
  bn_gelu_h<<<ceil_div(R_ * HC_, TPB), TPB, 0, stream>>>(pre, mean, var, gg, beg, hbf);

  // 9) fc2 GEMM: pre = hbf @ W2 + b2   [R x C]
  wmma_gemm_bf16<<<dim3(C_ / 32, R_ / 64), wblk, 0, stream>>>(hbf, w2pk, b2, pre, R_, C_, HC_);

  // 10) BN + residual -> out [B,C,H,W]
  run_stats(pre, C_);
  bn_final<<<ceil_div(R_ * C_, TPB), TPB, 0, stream>>>(pre, mean, var, g2, be2, x, out);
}
